// DecoderVMP_51015621542089
// MI455X (gfx1250) — matrix operand order repaired and checked by reading, not compiled, not run on hardware
//
#include <hip/hip_runtime.h>
#include <hip/hip_bf16.h>

// ---------------------------------------------------------------------------
// Problem constants (fixed by the reference)
// ---------------------------------------------------------------------------
constexpr int NL  = 6;
constexpr int HH  = 16;
constexpr int DD  = 1024;
constexpr int BB  = 2;
constexpr int TT  = 512;
constexpr int DH  = 64;
constexpr int BT  = BB * TT;          // 1024 rows for the linear GEMMs
constexpr int BTD = BB * TT * DD;     // 1,048,576
constexpr int BH  = BB * HH;          // 32 attention batches
constexpr long long BHTT  = (long long)BH * TT * TT;   // 8,388,608
constexpr long long BHTDH = (long long)BH * TT * DH;   // 1,048,576
constexpr float RD   = 32.0f;         // sqrt(1024)
constexpr float TOLF = 1e-5f;

typedef __bf16 bf16;
typedef __bf16 v16bf __attribute__((ext_vector_type(16)));
typedef __bf16 v8bf  __attribute__((ext_vector_type(8)));
typedef float  v8f   __attribute__((ext_vector_type(8)));
typedef int    v4i   __attribute__((ext_vector_type(4)));

// ---------------------------------------------------------------------------
// gfx1250 async global->LDS path (builtin confirmed present by round-1 probe)
// Signature: (v4i addrspace(1)* src, v4i addrspace(3)* dst, imm offset, imm cpol)
// ---------------------------------------------------------------------------
#if defined(__gfx1250__) && __has_builtin(__builtin_amdgcn_global_load_async_to_lds_b128) && __has_builtin(__builtin_amdgcn_s_wait_asynccnt)
#define USE_ASYNC_LDS 1
#else
#define USE_ASYNC_LDS 0
#endif

#if USE_ASYNC_LDS
static __device__ __forceinline__ void cp_async16(void* lds, const void* g) {
  __builtin_amdgcn_global_load_async_to_lds_b128(
      (__attribute__((address_space(1))) v4i*)g,
      (__attribute__((address_space(3))) v4i*)lds, 0, 0);
}
#endif

// ---------------------------------------------------------------------------
// WMMA bf16 GEMM:  D = alpha * (A1 @ B1^T  [+ A2 @ B2^T])
//   A*: row-major [M,K] bf16,  B*: row-major [N,K] bf16 (i.e. operand is B^T),
//   D : row-major [M,N] f32.  Batched via blockIdx.z with element strides.
// Tile: 128x64x32, 256 threads (8 waves), each wave -> 32x32 (4 WMMA accums).
// ---------------------------------------------------------------------------
template <int PAIRS>
__global__ __launch_bounds__(256) void gemm_bf16_wmma(
    const bf16* __restrict__ A1, const bf16* __restrict__ B1,
    const bf16* __restrict__ A2, const bf16* __restrict__ B2,
    float* __restrict__ Dm, int M, int N, int K,
    long long sA, long long sB, long long sD, float alpha) {
  constexpr int BM = 128, BN = 64, BK = 32;
  constexpr int LDT = BK + 8;  // padded LDS row stride (40 elems / 80B): conflict-free frags

  __shared__ bf16 As[2][PAIRS][BM * LDT];
  __shared__ bf16 Bs[2][PAIRS][BN * LDT];

  const int tid  = threadIdx.x;
  const int lane = tid & 31;
  const int wid  = tid >> 5;
  const int wm   = wid >> 1;      // 0..3  -> M offset wm*32
  const int wn   = wid & 1;       // 0..1  -> N offset wn*32
  const int l16  = lane & 15;
  const int half = lane >> 4;
  const int m0 = blockIdx.y * BM;
  const int n0 = blockIdx.x * BN;
  const long long zb = blockIdx.z;

  const bf16* Ap[2];
  const bf16* Bp[2];
  Ap[0] = A1 + zb * sA; Bp[0] = B1 + zb * sB;
  Ap[1] = (PAIRS == 2) ? (A2 + zb * sA) : Ap[0];
  Bp[1] = (PAIRS == 2) ? (B2 + zb * sB) : Bp[0];

  // global->LDS chunk map: A tile = 512 x 16B chunks (2/thread), B tile = 256 (1/thread)
  const int br = tid >> 2, bk = (tid & 3) * 8;

  v8f acc[2][2] = {};
  const int nst = K / BK;

#if USE_ASYNC_LDS
  auto issue = [&](int s) {
    const int buf = s & 1, k0 = s * BK;
    for (int p = 0; p < PAIRS; ++p) {
      for (int i = 0; i < 2; ++i) {
        int c = tid + i * 256, r = c >> 2, ko = (c & 3) * 8;
        cp_async16(&As[buf][p][r * LDT + ko], Ap[p] + (long long)(m0 + r) * K + k0 + ko);
      }
      cp_async16(&Bs[buf][p][br * LDT + bk], Bp[p] + (long long)(n0 + br) * K + k0 + bk);
    }
  };
  issue(0);
#else
  uint4 ra[PAIRS][2], rb[PAIRS];
  auto fetch = [&](int s) {
    const int k0 = s * BK;
    for (int p = 0; p < PAIRS; ++p) {
      for (int i = 0; i < 2; ++i) {
        int c = tid + i * 256, r = c >> 2, ko = (c & 3) * 8;
        ra[p][i] = *(const uint4*)(Ap[p] + (long long)(m0 + r) * K + k0 + ko);
      }
      rb[p] = *(const uint4*)(Bp[p] + (long long)(n0 + br) * K + k0 + bk);
    }
  };
  auto stash = [&](int s) {
    const int buf = s & 1;
    for (int p = 0; p < PAIRS; ++p) {
      for (int i = 0; i < 2; ++i) {
        int c = tid + i * 256, r = c >> 2, ko = (c & 3) * 8;
        *(uint4*)&As[buf][p][r * LDT + ko] = ra[p][i];
      }
      *(uint4*)&Bs[buf][p][br * LDT + bk] = rb[p];
    }
  };
  fetch(0);
#endif

  for (int s = 0; s < nst; ++s) {
    const int buf = s & 1;
#if USE_ASYNC_LDS
    if (s + 1 < nst) {
      issue(s + 1);
      __builtin_amdgcn_s_wait_asynccnt(3 * PAIRS);  // stage s landed, s+1 in flight
    } else {
      __builtin_amdgcn_s_wait_asynccnt(0);
    }
    __syncthreads();
#else
    stash(s);
    __syncthreads();
    if (s + 1 < nst) fetch(s + 1);  // overlap next-stage loads with compute
#endif
    for (int p = 0; p < PAIRS; ++p) {
      v16bf afr[2], bfr[2];
      for (int mt = 0; mt < 2; ++mt) {
        // ISA A-layout 16x32: lane<16 holds K 0..7 & 16..23; lane>=16 holds K 8..15 & 24..31
        const bf16* base = &As[buf][p][(wm * 32 + mt * 16 + l16) * LDT + half * 8];
        v8bf lo = *(const v8bf*)(base);
        v8bf hi = *(const v8bf*)(base + 16);
        afr[mt] = __builtin_shufflevector(lo, hi, 0,1,2,3,4,5,6,7,8,9,10,11,12,13,14,15);
      }
      for (int nt = 0; nt < 2; ++nt) {
        // B-layout 32x16 (matches A): lane<16 holds K 0..7 & 16..23 of col l16; lane>=16 holds K 8..15 & 24..31
        const bf16* base = &Bs[buf][p][(wn * 32 + nt * 16 + l16) * LDT + half * 8];
        v8bf lo = *(const v8bf*)(base);
        v8bf hi = *(const v8bf*)(base + 16);
        bfr[nt] = __builtin_shufflevector(lo, hi, 0,1,2,3,4,5,6,7,8,9,10,11,12,13,14,15);
      }
      for (int mt = 0; mt < 2; ++mt)
        for (int nt = 0; nt < 2; ++nt)
          acc[mt][nt] = __builtin_amdgcn_wmma_f32_16x16x32_bf16(
              false, afr[mt], false, bfr[nt], (short)0, acc[mt][nt], false, false);
    }
    __syncthreads();
  }

  float* Dp = Dm + zb * sD;
  for (int mt = 0; mt < 2; ++mt)
    for (int nt = 0; nt < 2; ++nt) {
      const int col = n0 + wn * 32 + nt * 16 + l16;
      for (int e = 0; e < 8; ++e) {
        const int row = m0 + wm * 32 + mt * 16 + e + half * 8;  // C/D VGPR layout
        Dp[(long long)row * N + col] = alpha * acc[mt][nt][e];
      }
    }
}

// ---------------------------------------------------------------------------
// Packing kernels (f32 -> bf16). MODE: 0 = a, 1 = a*a, 2 = a*a + b
// ---------------------------------------------------------------------------
template <int MODE>
static __device__ __forceinline__ float pk_f(float a, float b) {
  if (MODE == 0) return a;
  if (MODE == 1) return a * a;
  return a * a + b;
}

template <int MODE>
__global__ void pack_flat_k(const float* __restrict__ a, const float* __restrict__ b,
                            bf16* __restrict__ o, long long n) {
  long long i = (long long)blockIdx.x * 256 + threadIdx.x;
  if (i >= n) return;
  o[i] = (bf16)pk_f<MODE>(a[i], MODE == 2 ? b[i] : 0.0f);
}

// [B,T,D] f32 -> head-split bf16 [BH,T,DH] (TR=0) or [BH,DH,T] (TR=1)
template <int MODE, int TR>
__global__ void pack_split_k(const float* __restrict__ a, const float* __restrict__ b,
                             bf16* __restrict__ o) {
  long long i = (long long)blockIdx.x * 256 + threadIdx.x;
  if (i >= (long long)BTD) return;
  int dh = (int)(i % DH);
  int t  = (int)((i / DH) % TT);
  int bh = (int)(i / ((long long)DH * TT));
  int h = bh % HH, bb = bh / HH;
  long long src = ((long long)bb * TT + t) * DD + h * DH + dh;
  float y = pk_f<MODE>(a[src], MODE == 2 ? b[src] : 0.0f);
  long long dst = TR ? ((long long)bh * DH + dh) * TT + t : i;
  o[dst] = (bf16)y;
}

// [BH,T,DH] f32 -> [BH,DH,T] bf16 (transpose within batch)
template <int MODE>
__global__ void pack_tr_k(const float* __restrict__ a, const float* __restrict__ b,
                          bf16* __restrict__ o) {
  long long i = (long long)blockIdx.x * 256 + threadIdx.x;
  if (i >= BHTDH) return;
  int dh = (int)(i % DH);
  int t  = (int)((i / DH) % TT);
  int bh = (int)(i / ((long long)DH * TT));
  float y = pk_f<MODE>(a[i], MODE == 2 ? b[i] : 0.0f);
  o[((long long)bh * DH + dh) * TT + t] = (bf16)y;
}

// ---------------------------------------------------------------------------
// Elementwise / rowwise kernels
// ---------------------------------------------------------------------------
__global__ void add_pos_k(const float* __restrict__ xin, const float* __restrict__ pos,
                          float* __restrict__ x, float* __restrict__ vx) {
  long long i = (long long)blockIdx.x * 256 + threadIdx.x;
  if (i >= (long long)BTD) return;
  x[i]  = xin[i] + pos[i % ((long long)TT * DD)];
  vx[i] = 0.0f;
}

// x,vx [B,T,D] += merge(o,vo [BH,T,DH])
__global__ void resadd_merge_k(float* __restrict__ x, float* __restrict__ vx,
                               const float* __restrict__ o, const float* __restrict__ vo) {
  long long i = (long long)blockIdx.x * 256 + threadIdx.x;
  if (i >= (long long)BTD) return;
  int d = (int)(i % DD);
  int t = (int)((i / DD) % TT);
  int bb = (int)(i / ((long long)DD * TT));
  long long s = (((long long)(bb * HH + d / DH)) * TT + t) * DH + (d % DH);
  x[i] += o[s];
  vx[i] += vo[s];
}

__global__ void ffn_post_k(float* __restrict__ x, float* __restrict__ vx,
                           const float* __restrict__ y, const float* __restrict__ vy,
                           const float* __restrict__ bias, const float* __restrict__ vbias) {
  long long i = (long long)blockIdx.x * 256 + threadIdx.x;
  if (i >= (long long)BTD) return;
  int d = (int)(i % DD);
  float m = y[i] + bias[d];
  float vm = vy[i] + vbias[d];
  if (m > 0.0f) { x[i] += m; vx[i] += vm; }
}

__global__ void sigmoid_out_k(const float* __restrict__ x, const float* __restrict__ vx,
                              float* __restrict__ out) {
  long long i = (long long)blockIdx.x * 256 + threadIdx.x;
  if (i >= (long long)BTD) return;
  float s = 1.0f / (1.0f + __expf(-x[i]));
  float j = s * (1.0f - s);
  out[i] = s;
  out[(long long)BTD + i] = j * j * vx[i];
}

// LayerNorm-VMP, one 256-thread block per row of D=1024, in place
__global__ __launch_bounds__(256) void layernorm_k(
    float* __restrict__ x, float* __restrict__ vx,
    const float* __restrict__ g, const float* __restrict__ b,
    const float* __restrict__ vg, const float* __restrict__ vb) {
  __shared__ float red[256];
  const int t = threadIdx.x;
  float* xr = x + (long long)blockIdx.x * DD;
  float* vr = vx + (long long)blockIdx.x * DD;
  float xl[4];
  float s1 = 0.f, s2 = 0.f;
  for (int j = 0; j < 4; ++j) {
    float v = xr[t + j * 256];
    xl[j] = v; s1 += v; s2 += v * v;
  }
  red[t] = s1; __syncthreads();
  for (int o = 128; o; o >>= 1) { if (t < o) red[t] += red[t + o]; __syncthreads(); }
  float mu = red[0] / DD; __syncthreads();
  red[t] = s2; __syncthreads();
  for (int o = 128; o; o >>= 1) { if (t < o) red[t] += red[t + o]; __syncthreads(); }
  float var = red[0] / DD - mu * mu;
  float inv = rsqrtf(var + TOLF);
  for (int j = 0; j < 4; ++j) {
    int c = t + j * 256;
    float gg = g[c], vgg = vg[c];
    float xh = (xl[j] - mu) * inv;
    float vv = vr[c];
    xr[c] = gg * xh + b[c];
    vr[c] = vv * inv * inv * (gg * gg + vgg) + xh * xh * vgg + vb[c];
  }
}

// softmax_vmp + causal mask; one row per block; a/va pre-scaled by 1/RD, 1/RD^2
__global__ __launch_bounds__(256) void softmax_vmp_k(float* __restrict__ a,
                                                     float* __restrict__ va) {
  __shared__ float red[256];
  const long long r = blockIdx.x;        // 0 .. BH*T-1
  const int qi = (int)(r % TT);
  float* ar = a + r * TT;
  float* vr = va + r * TT;
  const int t = threadIdx.x;

  float av[2]; int cs[2];
  float mx = -INFINITY;
  for (int j = 0; j < 2; ++j) {
    int c = t + j * 256; cs[j] = c;
    av[j] = ar[c];
    if (c <= qi) mx = fmaxf(mx, av[j]);
  }
  red[t] = mx; __syncthreads();
  for (int o = 128; o; o >>= 1) { if (t < o) red[t] = fmaxf(red[t], red[t + o]); __syncthreads(); }
  mx = red[0]; __syncthreads();

  float e[2]; float sum = 0.f;
  for (int j = 0; j < 2; ++j) {
    e[j] = (cs[j] <= qi) ? __expf(av[j] - mx) : 0.0f;
    sum += e[j];
  }
  red[t] = sum; __syncthreads();
  for (int o = 128; o; o >>= 1) { if (t < o) red[t] += red[t + o]; __syncthreads(); }
  const float inv = 1.0f / red[0]; __syncthreads();

  float sj[2], vav[2]; float tac = 0.f;
  for (int j = 0; j < 2; ++j) {
    sj[j] = e[j] * inv;
    vav[j] = vr[cs[j]];
    tac += sj[j] * sj[j] * vav[j];   // masked: sj == 0
  }
  red[t] = tac; __syncthreads();
  for (int o = 128; o; o >>= 1) { if (t < o) red[t] += red[t + o]; __syncthreads(); }
  const float tt = red[0];

  for (int j = 0; j < 2; ++j) {
    float vs = sj[j] * sj[j] * (tt + (1.0f - 2.0f * sj[j]) * vav[j]);
    ar[cs[j]] = sj[j];
    vr[cs[j]] = (cs[j] <= qi) ? vs : 0.0f;
  }
}

// ---------------------------------------------------------------------------
// Host orchestration
// ---------------------------------------------------------------------------
extern "C" void kernel_launch(void* const* d_in, const int* in_sizes, int n_in,
                              void* d_out, int out_size, void* d_ws, size_t ws_size,
                              hipStream_t stream) {
  (void)in_sizes; (void)n_in; (void)out_size; (void)ws_size;
  const float* in_x   = (const float*)d_in[0];
  const float* in_k   = (const float*)d_in[1];
  const float* in_vk  = (const float*)d_in[2];
  const float* in_v   = (const float*)d_in[3];
  const float* in_vv  = (const float*)d_in[4];
  const float* in_pos = (const float*)d_in[5];
  const float* Wq1 = (const float*)d_in[6],  *Vq1 = (const float*)d_in[7];
  const float* Wk1 = (const float*)d_in[8],  *Vk1 = (const float*)d_in[9];
  const float* Wv1 = (const float*)d_in[10], *Vv1 = (const float*)d_in[11];
  const float* Wq2 = (const float*)d_in[12], *Vq2 = (const float*)d_in[13];
  const float* Wfc = (const float*)d_in[14], *Vfc = (const float*)d_in[15];
  const float* bfc = (const float*)d_in[16], *Vbfc = (const float*)d_in[17];
  const float* g1 = (const float*)d_in[18], *b1 = (const float*)d_in[19];
  const float* vg1 = (const float*)d_in[20], *vb1 = (const float*)d_in[21];
  const float* g2 = (const float*)d_in[22], *b2 = (const float*)d_in[23];
  const float* vg2 = (const float*)d_in[24], *vb2 = (const float*)d_in[25];
  const float* g3 = (const float*)d_in[26], *b3 = (const float*)d_in[27];
  const float* vg3 = (const float*)d_in[28], *vb3 = (const float*)d_in[29];

  // workspace carve (bump allocator, 256B aligned)
  char* base = (char*)d_ws;
  size_t off = 0;
  auto carve = [&](size_t bytes) -> void* {
    void* p = base + off;
    off += (bytes + 255) & ~(size_t)255;
    return p;
  };
  float* f_x   = (float*)carve((size_t)BTD * 4);
  float* f_vx  = (float*)carve((size_t)BTD * 4);
  float* f_q   = (float*)carve((size_t)BTD * 4);
  float* f_vq  = (float*)carve((size_t)BTD * 4);
  float* f_k   = (float*)carve((size_t)BTD * 4);
  float* f_vk  = (float*)carve((size_t)BTD * 4);
  float* f_vm  = (float*)carve((size_t)BTD * 4);   // value mean
  float* f_vv  = (float*)carve((size_t)BTD * 4);   // value var
  float* f_ao  = (float*)carve((size_t)BHTDH * 4); // attn out [BH,T,DH]
  float* f_vao = (float*)carve((size_t)BHTDH * 4);
  float* f_sc  = (float*)carve((size_t)BHTT * 4);  // scores mean [BH,T,T]
  float* f_vsc = (float*)carve((size_t)BHTT * 4);  // scores var
  bf16* b_x    = (bf16*)carve((size_t)BTD * 2);
  bf16* b_vx   = (bf16*)carve((size_t)BTD * 2);
  bf16* b_xsq  = (bf16*)carve((size_t)BTD * 2);
  bf16* b_w1   = (bf16*)carve((size_t)DD * DD * 2);
  bf16* b_w2   = (bf16*)carve((size_t)DD * DD * 2);
  bf16* b_w3   = (bf16*)carve((size_t)DD * DD * 2);
  bf16* b_q    = (bf16*)carve((size_t)BHTDH * 2);
  bf16* b_qsq  = (bf16*)carve((size_t)BHTDH * 2);
  bf16* b_vqh  = (bf16*)carve((size_t)BHTDH * 2);
  bf16* b_k    = (bf16*)carve((size_t)BHTDH * 2);
  bf16* b_vkh  = (bf16*)carve((size_t)BHTDH * 2);
  bf16* b_kvs  = (bf16*)carve((size_t)BHTDH * 2);  // k^2 + vk
  bf16* b_vT   = (bf16*)carve((size_t)BHTDH * 2);  // v^T        [BH,DH,T]
  bf16* b_vvT  = (bf16*)carve((size_t)BHTDH * 2);  // (v^2+vv)^T
  bf16* b_varT = (bf16*)carve((size_t)BHTDH * 2);  // vv^T
  bf16* b_s1   = (bf16*)carve((size_t)BHTT * 2);
  bf16* b_s2   = (bf16*)carve((size_t)BHTT * 2);

  const dim3 blk(256);
  auto g1d = [](long long n) { return dim3((unsigned)((n + 255) / 256)); };

  auto gemm1 = [&](const bf16* A, const bf16* Bm, float* Dp, int M, int N, int K,
                   long long sA, long long sB, long long sD, int batch, float alpha) {
    dim3 grid(N / 64, M / 128, batch);
    gemm_bf16_wmma<1><<<grid, blk, 0, stream>>>(A, Bm, nullptr, nullptr, Dp,
                                                M, N, K, sA, sB, sD, alpha);
  };
  auto gemm2 = [&](const bf16* A1p, const bf16* B1p, const bf16* A2p, const bf16* B2p,
                   float* Dp, int M, int N, int K,
                   long long sA, long long sB, long long sD, int batch, float alpha) {
    dim3 grid(N / 64, M / 128, batch);
    gemm_bf16_wmma<2><<<grid, blk, 0, stream>>>(A1p, B1p, A2p, B2p, Dp,
                                                M, N, K, sA, sB, sD, alpha);
  };

  auto packX = [&]() {
    pack_flat_k<0><<<g1d(BTD), blk, 0, stream>>>(f_x, nullptr, b_x, BTD);
    pack_flat_k<0><<<g1d(BTD), blk, 0, stream>>>(f_vx, nullptr, b_vx, BTD);
    pack_flat_k<1><<<g1d(BTD), blk, 0, stream>>>(f_x, nullptr, b_xsq, BTD);
  };

  // linear_vmp: ym = x@W^T ; yv = vx@(W^2+VW)^T + x^2@VW^T
  auto linear = [&](const float* W, const float* VW, float* ym, float* yv) {
    const long long nw = (long long)DD * DD;
    pack_flat_k<0><<<g1d(nw), blk, 0, stream>>>(W, nullptr, b_w1, nw);
    pack_flat_k<2><<<g1d(nw), blk, 0, stream>>>(W, VW, b_w2, nw);
    pack_flat_k<0><<<g1d(nw), blk, 0, stream>>>(VW, nullptr, b_w3, nw);
    gemm1(b_x, b_w1, ym, BT, DD, DD, 0, 0, 0, 1, 1.0f);
    gemm2(b_vx, b_w2, b_xsq, b_w3, yv, BT, DD, DD, 0, 0, 0, 1, 1.0f);
  };

  // attention: q/vq in f_q/f_vq ([B,T,D]); k/v either [B,T,D] (self) or [B,H,T,DH] (cross)
  auto attend = [&](bool cross, const float* km, const float* kv,
                    const float* vm, const float* vv) {
    pack_split_k<0, 0><<<g1d(BTD), blk, 0, stream>>>(f_q, nullptr, b_q);
    pack_split_k<1, 0><<<g1d(BTD), blk, 0, stream>>>(f_q, nullptr, b_qsq);
    pack_split_k<0, 0><<<g1d(BTD), blk, 0, stream>>>(f_vq, nullptr, b_vqh);
    if (cross) {
      pack_flat_k<0><<<g1d(BHTDH), blk, 0, stream>>>(km, nullptr, b_k, BHTDH);
      pack_flat_k<0><<<g1d(BHTDH), blk, 0, stream>>>(kv, nullptr, b_vkh, BHTDH);
      pack_flat_k<2><<<g1d(BHTDH), blk, 0, stream>>>(km, kv, b_kvs, BHTDH);
    } else {
      pack_split_k<0, 0><<<g1d(BTD), blk, 0, stream>>>(km, nullptr, b_k);
      pack_split_k<0, 0><<<g1d(BTD), blk, 0, stream>>>(kv, nullptr, b_vkh);
      pack_split_k<2, 0><<<g1d(BTD), blk, 0, stream>>>(km, kv, b_kvs);
    }
    const long long sQK = (long long)TT * DH, sS = (long long)TT * TT;
    gemm1(b_q, b_k, f_sc, TT, TT, DH, sQK, sQK, sS, BH, 1.0f / RD);
    gemm2(b_vqh, b_kvs, b_qsq, b_vkh, f_vsc, TT, TT, DH, sQK, sQK, sS, BH,
          1.0f / (RD * RD));
    softmax_vmp_k<<<dim3(BH * TT), blk, 0, stream>>>(f_sc, f_vsc);
    if (cross) {
      pack_tr_k<0><<<g1d(BHTDH), blk, 0, stream>>>(vm, nullptr, b_vT);
      pack_tr_k<0><<<g1d(BHTDH), blk, 0, stream>>>(vv, nullptr, b_varT);
      pack_tr_k<2><<<g1d(BHTDH), blk, 0, stream>>>(vm, vv, b_vvT);
    } else {
      pack_split_k<0, 1><<<g1d(BTD), blk, 0, stream>>>(vm, nullptr, b_vT);
      pack_split_k<0, 1><<<g1d(BTD), blk, 0, stream>>>(vv, nullptr, b_varT);
      pack_split_k<2, 1><<<g1d(BTD), blk, 0, stream>>>(vm, vv, b_vvT);
    }
    const long long sVT = (long long)DH * TT;
    pack_flat_k<0><<<g1d(BHTT), blk, 0, stream>>>(f_sc, nullptr, b_s1, BHTT);
    gemm1(b_s1, b_vT, f_ao, TT, DH, TT, sS, sVT, sQK, BH, 1.0f);
    pack_flat_k<0><<<g1d(BHTT), blk, 0, stream>>>(f_vsc, nullptr, b_s1, BHTT);
    pack_flat_k<1><<<g1d(BHTT), blk, 0, stream>>>(f_sc, nullptr, b_s2, BHTT);
    gemm2(b_s1, b_vvT, b_s2, b_varT, f_vao, TT, DH, TT, sS, sVT, sQK, BH, 1.0f);
    resadd_merge_k<<<g1d(BTD), blk, 0, stream>>>(f_x, f_vx, f_ao, f_vao);
  };

  // ---------------- forward pass ----------------
  add_pos_k<<<g1d(BTD), blk, 0, stream>>>(in_x, in_pos, f_x, f_vx);

  for (int i = 0; i < NL; ++i) {
    const size_t wO = (size_t)i * DD * DD;
    const size_t dO = (size_t)i * DD;
    if (i) {
      const size_t pO = (size_t)(i - 1) * DD;
      layernorm_k<<<dim3(BT), blk, 0, stream>>>(f_x, f_vx, g1 + pO, b1 + pO,
                                                vg1 + pO, vb1 + pO);
    }
    // masked self-attention
    packX();
    linear(Wq1 + wO, Vq1 + wO, f_q, f_vq);
    linear(Wk1 + wO, Vk1 + wO, f_k, f_vk);
    linear(Wv1 + wO, Vv1 + wO, f_vm, f_vv);
    attend(false, f_k, f_vk, f_vm, f_vv);
    layernorm_k<<<dim3(BT), blk, 0, stream>>>(f_x, f_vx, g2 + dO, b2 + dO,
                                              vg2 + dO, vb2 + dO);
    // cross-attention with encoder k/v
    packX();
    linear(Wq2 + wO, Vq2 + wO, f_q, f_vq);
    attend(true, in_k, in_vk, in_v, in_vv);
    layernorm_k<<<dim3(BT), blk, 0, stream>>>(f_x, f_vx, g3 + dO, b3 + dO,
                                              vg3 + dO, vb3 + dO);
    // FFN
    packX();
    linear(Wfc + wO, Vfc + wO, f_q, f_vq);
    ffn_post_k<<<g1d(BTD), blk, 0, stream>>>(f_x, f_vx, f_q, f_vq, bfc + dO, Vbfc + dO);
  }
  sigmoid_out_k<<<g1d(BTD), blk, 0, stream>>>(f_x, f_vx, (float*)d_out);
}
